// Attention_28243704938593
// MI455X (gfx1250) — matrix operand order repaired and checked by reading, not compiled, not run on hardware
//
#include <hip/hip_runtime.h>
#include <hip/hip_bf16.h>

// ---- CDNA5 (gfx1250) wave32 WMMA attention pipeline -------------------------
// Matmuls: v_wmma_f32_16x16x32_bf16 (fp32 accumulate).
// GEMM staging: double-buffered, A tile via global_load_async_to_lds_b128
// (ASYNCcnt path) overlapped with WMMA compute on the other buffer.

typedef __attribute__((ext_vector_type(16))) __bf16 bf16x16;
typedef __attribute__((ext_vector_type(8)))  float  f32x8;
typedef __attribute__((ext_vector_type(4)))  int    i32x4;
typedef unsigned short u16;
typedef unsigned int   u32;

#define B_    2
#define T_    3136
#define DIM_  768
#define H_    12
#define DH_   64
#define NP_   196
#define ROWS_ (B_*T_)      // 6272
#define QKVN_ (3*DIM_)     // 2304

__device__ __forceinline__ u16 f2bf(float f) {
  u32 u = __float_as_uint(f);
  u32 r = u + 0x7FFFu + ((u >> 16) & 1u);   // round-to-nearest-even
  return (u16)(r >> 16);
}

union AF {            // one WMMA 16-bit operand fragment: 16 bf16 = 32 bytes
  i32x4  q[2];
  bf16x16 v;
  u16    u[16];
};

__device__ __forceinline__ f32x8 wmma_bf16(bf16x16 a, bf16x16 b, f32x8 c) {
  return __builtin_amdgcn_wmma_f32_16x16x32_bf16(
      /*neg_a=*/false, a, /*neg_b=*/false, b,
      /*c_mod=*/(short)0, c, /*reuse_a=*/false, /*reuse_b=*/false);
}

// Generic LDS pointer -> wave-relative DS byte offset (low 32 bits of the
// flat LDS-aperture address ARE the DS offset on amdgcn).
__device__ __forceinline__ unsigned lds_off(const void* p) {
  return (unsigned)(size_t)p;
}

// Async copy of 64 contiguous bytes global->LDS per lane (4x b128, ASYNCcnt).
__device__ __forceinline__ void async_copy_row64(unsigned ldsoff, const u16* g) {
  unsigned long long ga = (unsigned long long)(size_t)g;
  asm volatile("global_load_async_to_lds_b128 %0, %1, off\n\t"
               "global_load_async_to_lds_b128 %0, %1, off offset:16\n\t"
               "global_load_async_to_lds_b128 %0, %1, off offset:32\n\t"
               "global_load_async_to_lds_b128 %0, %1, off offset:48"
               :: "v"(ldsoff), "v"(ga) : "memory");
}

__device__ __forceinline__ void wait_async0() {
  asm volatile("s_wait_asynccnt 0" ::: "memory");
}

// ---------------------------------------------------------------------------
// Kernel 1: fp32 -> bf16 weight conversion (grid-stride)
// ---------------------------------------------------------------------------
__global__ void cvt_bf16_kernel(const float* __restrict__ src,
                                u16* __restrict__ dst, int n) {
  int i = blockIdx.x * blockDim.x + threadIdx.x;
  int stride = gridDim.x * blockDim.x;
  for (; i < n; i += stride) dst[i] = f2bf(src[i]);
}

// ---------------------------------------------------------------------------
// Kernel 2: LayerNorm, one wave32 per row of 768, output bf16
// ---------------------------------------------------------------------------
__global__ void ln_kernel(const float* __restrict__ x,
                          const float* __restrict__ gamma,
                          const float* __restrict__ beta,
                          u16* __restrict__ xn) {
  int wave = threadIdx.x >> 5;
  int lane = threadIdx.x & 31;
  int row  = blockIdx.x * 8 + wave;           // 784 blocks * 8 waves = 6272
  const float* xr = x + (size_t)row * DIM_;
  float v[24];
  float s = 0.f, ss = 0.f;
#pragma unroll
  for (int j = 0; j < 24; ++j) {
    float f = xr[lane + j * 32];
    v[j] = f; s += f; ss += f * f;
  }
#pragma unroll
  for (int m = 16; m >= 1; m >>= 1) {         // full-wave32 reduction
    s  += __shfl_xor(s,  m);
    ss += __shfl_xor(ss, m);
  }
  float mean = s * (1.0f / DIM_);
  float var  = ss * (1.0f / DIM_) - mean * mean;
  float rs   = rsqrtf(var + 1e-5f);
  u16* yr = xn + (size_t)row * DIM_;
#pragma unroll
  for (int j = 0; j < 24; ++j) {
    int c = lane + j * 32;
    yr[c] = f2bf((v[j] - mean) * rs * gamma[c] + beta[c]);
  }
}

// ---------------------------------------------------------------------------
// Shared GEMM mainloop: 128x64 tile per 128-thread block (4 waves),
// K stepped by 32, DOUBLE-BUFFERED LDS. Each wave owns 32 rows x 64 cols
// -> 8 WMMAs per K-step. While the WMMAs for buffer `cur` run, the async
// engine fills buffer `nxt` (A tile) and the B tile for `nxt` is fetched
// from global; its transpose-store into LDS is issued after the WMMAs.
// One barrier per K-step.
// ---------------------------------------------------------------------------
#define AS_SZ (128 * 32)
#define BS_SZ (64 * 32)

__device__ __forceinline__ void gemm_mainloop(
    const u16* __restrict__ A, const u16* __restrict__ Bw,
    int lda, int ldb, int tileM, int tileN, int K,
    u16* As, u16* Bs, f32x8 acc[2][4]) {
  const int tid  = threadIdx.x;
  const int wv   = tid >> 5;
  const int lane = tid & 31;
  const int g    = lane >> 4;
  const int nn   = lane & 15;
  const int kk   = tid >> 2;          // B-tile staging row (0..31)
  const int part = tid & 3;           // B-tile staging col group
  f32x8 zero = {0.f, 0.f, 0.f, 0.f, 0.f, 0.f, 0.f, 0.f};
#pragma unroll
  for (int s = 0; s < 2; ++s)
#pragma unroll
    for (int j = 0; j < 4; ++j) acc[s][j] = zero;

  const u16* arow_g = A + (size_t)(tileM + tid) * lda;   // this thread's A row
  const u16* brow_g = Bw + (size_t)kk * ldb + tileN + part * 16;

  union BT { i32x4 q[2]; u16 u[16]; };

  // ---- prologue: stage tile 0 into buffer 0
  async_copy_row64(lds_off(As + tid * 32), arow_g);
  {
    BT bt;
    bt.q[0] = ((const i32x4*)brow_g)[0];
    bt.q[1] = ((const i32x4*)brow_g)[1];
#pragma unroll
    for (int i = 0; i < 16; ++i) Bs[(part * 16 + i) * 32 + kk] = bt.u[i];
  }

  const int NK = K / 32;
  for (int ki = 0; ki < NK; ++ki) {
    const int cur = ki & 1;
    u16* Asc = As + cur * AS_SZ;
    u16* Bsc = Bs + cur * BS_SZ;
    u16* Asn = As + (1 - cur) * AS_SZ;
    u16* Bsn = Bs + (1 - cur) * BS_SZ;

    wait_async0();        // tile `ki`'s A asyncs (issued last iter) landed
    __syncthreads();      // + everyone's B transpose-stores visible

    const bool more = (ki + 1) < NK;
    BT btn;
    if (more) {
      // kick next tile's fills while we compute on `cur`
      async_copy_row64(lds_off(Asn + tid * 32), arow_g + (ki + 1) * 32);
      const u16* bsrc = brow_g + (size_t)(ki + 1) * 32 * ldb;
      btn.q[0] = ((const i32x4*)bsrc)[0];
      btn.q[1] = ((const i32x4*)bsrc)[1];
      if (ki + 2 < NK)
        __builtin_prefetch(brow_g + (size_t)(ki + 2) * 32 * ldb, 0, 0);
    }

    // ---- 8 WMMAs on buffer `cur`
    AF a[2];
#pragma unroll
    for (int s = 0; s < 2; ++s) {
      int ar = wv * 32 + s * 16 + nn;
      a[s].q[0] = *(const i32x4*)(Asc + ar * 32 + g * 16);
      a[s].q[1] = *(const i32x4*)(Asc + ar * 32 + g * 16 + 8);
    }
#pragma unroll
    for (int j = 0; j < 4; ++j) {
      AF b;
      const u16* bp = Bsc + (j * 16 + nn) * 32 + g * 16;
      b.q[0] = ((const i32x4*)bp)[0];
      b.q[1] = ((const i32x4*)bp)[1];
#pragma unroll
      for (int s = 0; s < 2; ++s) acc[s][j] = wmma_bf16(a[s].v, b.v, acc[s][j]);
    }

    // ---- transpose-store next B tile into the other buffer
    if (more) {
#pragma unroll
      for (int i = 0; i < 16; ++i) Bsn[(part * 16 + i) * 32 + kk] = btn.u[i];
    }
  }
}

// ---------------------------------------------------------------------------
// Kernel 3: QKV GEMM  [6272,768] x [768,2304] -> q,k,v bf16 [B,H,T,64]
// q is pre-scaled by DH^-0.5 = 0.125
// ---------------------------------------------------------------------------
__global__ void qkv_gemm_kernel(const u16* __restrict__ xn,
                                const u16* __restrict__ wq,
                                u16* __restrict__ qb, u16* __restrict__ kb,
                                u16* __restrict__ vb) {
  __shared__ u16 As[2 * AS_SZ];
  __shared__ u16 Bs[2 * BS_SZ];
  int tileM = blockIdx.y * 128, tileN = blockIdx.x * 64;
  f32x8 acc[2][4];
  gemm_mainloop(xn, wq, DIM_, QKVN_, tileM, tileN, DIM_, As, Bs, acc);

  int tid = threadIdx.x, wv = tid >> 5, lane = tid & 31;
  int g = lane >> 4, nn = lane & 15;
#pragma unroll
  for (int j = 0; j < 4; ++j) {
    int col   = tileN + j * 16 + nn;
    int which = col / DIM_;                  // 0:q 1:k 2:v (uniform per j-subtile)
    int rem   = col - which * DIM_;
    int hh    = rem >> 6, d = rem & 63;
    u16* dst  = (which == 0) ? qb : ((which == 1) ? kb : vb);
    float sc  = (which == 0) ? 0.125f : 1.0f;
#pragma unroll
    for (int s = 0; s < 2; ++s)
#pragma unroll
      for (int r = 0; r < 8; ++r) {
        int row = tileM + wv * 32 + s * 16 + 8 * g + r;   // global row over B*T
        int bq  = row / T_;
        int t   = row - bq * T_;
        dst[((size_t)(bq * H_ + hh) * T_ + t) * DH_ + d] = f2bf(acc[s][j][r] * sc);
      }
  }
}

// ---------------------------------------------------------------------------
// Kernel 4: block-causal flash attention.
// One wave32 per (b, h, 16-row q-tile); 32 keys per inner step.
// No block barriers (waves have divergent causal trip counts) --
// wave-internal LDS ordering via s_wait_dscnt.
// ---------------------------------------------------------------------------
__global__ void attn_kernel(const u16* __restrict__ qb,
                            const u16* __restrict__ kb,
                            const u16* __restrict__ vb,
                            u16* __restrict__ ao) {
  __shared__ u16 lds[4 * (16 * 32 + 64 * 32)];   // per-wave: P(16x32) + Vt(64x32)
  int wib  = threadIdx.x >> 5;
  int lane = threadIdx.x & 31;
  int g = lane >> 4, nn = lane & 15;
  u16* Pt = lds + wib * (16 * 32 + 64 * 32);
  u16* Vt = Pt + 16 * 32;

  int wid = blockIdx.x * 4 + wib;          // 0..4703
  int qt  = wid % 196;
  int bh  = wid / 196;                     // b*H + h
  int q0  = qt * 16;

  const u16* Q  = qb + (size_t)bh * T_ * DH_;
  const u16* Kp = kb + (size_t)bh * T_ * DH_;
  const u16* Vp = vb + (size_t)bh * T_ * DH_;

  // Q fragments (2 d-chunks of 32), held in registers for the whole k-loop
  AF qa[2];
  {
    const u16* qr = Q + (size_t)(q0 + nn) * DH_;
#pragma unroll
    for (int dc = 0; dc < 2; ++dc) {
      qa[dc].q[0] = *(const i32x4*)(qr + dc * 32 + g * 16);
      qa[dc].q[1] = *(const i32x4*)(qr + dc * 32 + g * 16 + 8);
    }
  }

  int qfr[8];
#pragma unroll
  for (int r = 0; r < 8; ++r) qfr[r] = (q0 + 8 * g + r) / NP_;
  int kmax = ((q0 + 15) / NP_ + 1) * NP_;  // exclusive causal key bound

  f32x8 zero = {0.f, 0.f, 0.f, 0.f, 0.f, 0.f, 0.f, 0.f};
  f32x8 o[4];
  o[0] = o[1] = o[2] = o[3] = zero;
  float mrow[8], lrow[8];
#pragma unroll
  for (int r = 0; r < 8; ++r) { mrow[r] = -3.0e38f; lrow[r] = 0.f; }

  for (int kt = 0; kt < kmax; kt += 32) {
    // stage V tile transposed: Vt[d][key], lane handles key = kt+lane
    {
      const u16* vr = Vp + (size_t)(kt + lane) * DH_;
      union { i32x4 q[8]; u16 u[64]; } t;
#pragma unroll
      for (int i = 0; i < 8; ++i) t.q[i] = ((const i32x4*)vr)[i];
#pragma unroll
      for (int d = 0; d < 64; ++d) Vt[d * 32 + lane] = t.u[d];
    }

    // S = Q*K^T for two 16-key halves; K^T B-fragments come straight from
    // global (K is [T,64] so d is contiguous -> two b128 loads per fragment)
    f32x8 s[2];
#pragma unroll
    for (int hh = 0; hh < 2; ++hh) {
      s[hh] = zero;
#pragma unroll
      for (int dc = 0; dc < 2; ++dc) {
        AF kf;
        const u16* kr = Kp + (size_t)(kt + hh * 16 + nn) * DH_ + dc * 32 + g * 16;
        kf.q[0] = ((const i32x4*)kr)[0];
        kf.q[1] = ((const i32x4*)kr)[1];
        s[hh] = wmma_bf16(qa[dc].v, kf.v, s[hh]);
      }
    }

    // analytic block-causal mask: frame(q) >= frame(k)
#pragma unroll
    for (int hh = 0; hh < 2; ++hh) {
      int key = kt + hh * 16 + nn;
      int kfr = key / NP_;
#pragma unroll
      for (int r = 0; r < 8; ++r) {
        if (!((key < kmax) && (kfr <= qfr[r]))) s[hh][r] = -1.0e30f;
      }
    }

    // online softmax (row reductions across the 16-lane half-wave)
    float p0[8], p1[8], scl[8];
#pragma unroll
    for (int r = 0; r < 8; ++r) {
      float mx = fmaxf(s[0][r], s[1][r]);
#pragma unroll
      for (int m = 8; m >= 1; m >>= 1) mx = fmaxf(mx, __shfl_xor(mx, m));
      float mn = fmaxf(mrow[r], mx);
      scl[r] = __expf(mrow[r] - mn);
      p0[r]  = __expf(s[0][r] - mn);
      p1[r]  = __expf(s[1][r] - mn);
      float sum = p0[r] + p1[r];
#pragma unroll
      for (int m = 8; m >= 1; m >>= 1) sum += __shfl_xor(sum, m);
      lrow[r] = lrow[r] * scl[r] + sum;
      mrow[r] = mn;
    }
#pragma unroll
    for (int j = 0; j < 4; ++j)
#pragma unroll
      for (int r = 0; r < 8; ++r) o[j][r] *= scl[r];

    // P (C/D layout) -> LDS -> reload in A-fragment layout
#pragma unroll
    for (int r = 0; r < 8; ++r) {
      Pt[(8 * g + r) * 32 + nn]      = f2bf(p0[r]);
      Pt[(8 * g + r) * 32 + 16 + nn] = f2bf(p1[r]);
    }
    asm volatile("s_wait_dscnt 0" ::: "memory");   // wave-internal LDS ordering

    AF pa;
    pa.q[0] = *(const i32x4*)(Pt + nn * 32 + g * 16);
    pa.q[1] = *(const i32x4*)(Pt + nn * 32 + g * 16 + 8);
#pragma unroll
    for (int j = 0; j < 4; ++j) {
      AF vf;
      const u16* vpp = Vt + (j * 16 + nn) * 32 + g * 16;
      vf.q[0] = ((const i32x4*)vpp)[0];
      vf.q[1] = ((const i32x4*)vpp)[1];
      o[j] = wmma_bf16(pa.v, vf.v, o[j]);
    }
    asm volatile("" ::: "memory");  // keep next-iter DS stores below these loads
  }

  // epilogue: normalize and store bf16 [B,T,H*DH]
  int b = bh / H_, h = bh - b * H_;
#pragma unroll
  for (int j = 0; j < 4; ++j)
#pragma unroll
    for (int r = 0; r < 8; ++r) {
      int row = q0 + 8 * g + r;
      ao[((size_t)(b * T_ + row)) * DIM_ + h * DH_ + j * 16 + nn] =
          f2bf(o[j][r] / lrow[r]);
    }
}

// ---------------------------------------------------------------------------
// Kernel 5: output projection [6272,768] x [768,768] + bias -> fp32
// ---------------------------------------------------------------------------
__global__ void out_gemm_kernel(const u16* __restrict__ ao,
                                const u16* __restrict__ wo,
                                const float* __restrict__ bias,
                                float* __restrict__ out) {
  __shared__ u16 As[2 * AS_SZ];
  __shared__ u16 Bs[2 * BS_SZ];
  int tileM = blockIdx.y * 128, tileN = blockIdx.x * 64;
  f32x8 acc[2][4];
  gemm_mainloop(ao, wo, DIM_, DIM_, tileM, tileN, DIM_, As, Bs, acc);

  int tid = threadIdx.x, wv = tid >> 5, lane = tid & 31;
  int g = lane >> 4, nn = lane & 15;
#pragma unroll
  for (int j = 0; j < 4; ++j) {
    int col = tileN + j * 16 + nn;
    float bv = bias[col];
#pragma unroll
    for (int s = 0; s < 2; ++s)
#pragma unroll
      for (int r = 0; r < 8; ++r) {
        int row = tileM + wv * 32 + s * 16 + 8 * g + r;
        out[(size_t)row * DIM_ + col] = acc[s][j][r] + bv;
      }
  }
}

// ---------------------------------------------------------------------------
extern "C" void kernel_launch(void* const* d_in, const int* in_sizes, int n_in,
                              void* d_out, int out_size, void* d_ws, size_t ws_size,
                              hipStream_t stream) {
  (void)in_sizes; (void)n_in; (void)out_size; (void)ws_size;
  const float* x      = (const float*)d_in[0];
  const float* gamma  = (const float*)d_in[1];
  const float* beta   = (const float*)d_in[2];
  const float* w_qkv  = (const float*)d_in[3];
  const float* w_out  = (const float*)d_in[4];
  const float* b_out  = (const float*)d_in[5];
  // d_in[6] = mask: unused, block-causal structure is computed analytically
  float* out = (float*)d_out;

  // workspace layout (bf16), total ~50.4 MB
  u16* ws  = (u16*)d_ws;
  u16* xn  = ws;  ws += (size_t)ROWS_ * DIM_;          // 6272*768
  u16* wqb = ws;  ws += (size_t)DIM_ * QKVN_;          // 768*2304
  u16* wob = ws;  ws += (size_t)DIM_ * DIM_;           // 768*768
  u16* qb  = ws;  ws += (size_t)B_ * H_ * T_ * DH_;
  u16* kb  = ws;  ws += (size_t)B_ * H_ * T_ * DH_;
  u16* vb  = ws;  ws += (size_t)B_ * H_ * T_ * DH_;
  u16* ao  = ws;  // 6272*768

  cvt_bf16_kernel<<<512, 256, 0, stream>>>(w_qkv, wqb, DIM_ * QKVN_);
  cvt_bf16_kernel<<<256, 256, 0, stream>>>(w_out, wob, DIM_ * DIM_);
  ln_kernel<<<ROWS_ / 8, 256, 0, stream>>>(x, gamma, beta, xn);
  qkv_gemm_kernel<<<dim3(QKVN_ / 64, ROWS_ / 128), 128, 0, stream>>>(xn, wqb, qb, kb, vb);
  attn_kernel<<<(B_ * H_ * (T_ / 16)) / 4, 128, 0, stream>>>(qb, kb, vb, ao);
  out_gemm_kernel<<<dim3(DIM_ / 64, ROWS_ / 128), 128, 0, stream>>>(ao, wob, b_out, out);
}
